// WindowAttention_50431505990288
// MI455X (gfx1250) — matrix operand order repaired and checked by reading, not compiled, not run on hardware
//
#include <hip/hip_runtime.h>
#include <hip/hip_bf16.h>

// MI455X / gfx1250 fused Swin window-attention.
// Precision: f16 WMMA (V_WMMA_F32_16X16X32_F16), f32 accumulate.
// One workgroup (256 threads = 8 wave32) handles one window (49 tokens -> padded 64).
// Wave w == head w for the attention phase (HEADS == 8 == waves/WG).

typedef _Float16 h4_t  __attribute__((ext_vector_type(4)));
typedef _Float16 h8_t  __attribute__((ext_vector_type(8)));
typedef _Float16 h16_t __attribute__((ext_vector_type(16)));
typedef float    f8_t  __attribute__((ext_vector_type(8)));

#define WMMA16(a, b, c) \
  __builtin_amdgcn_wmma_f32_16x16x32_f16(false, (a), false, (b), (short)0, (c), false, false)

__device__ __forceinline__ h16_t frag_ld(const _Float16* p, int o0, int o1) {
  h8_t a = *(const h8_t*)(p + o0);
  h8_t b = *(const h8_t*)(p + o1);
  return __builtin_shufflevector(a, b, 0, 1, 2, 3, 4, 5, 6, 7, 8, 9, 10, 11, 12, 13, 14, 15);
}

// A fragment (16x32 f16, M x K). lane: m = R + (l&15); K chunks at lh*8 and 16+lh*8.
__device__ __forceinline__ h16_t fragA(const _Float16* p, int S, int R, int KB, int l15, int lh) {
  int o = (R + l15) * S + KB + lh * 8;
  return frag_ld(p, o, o + 16);
}
// B fragment (32x16 f16, K x N) from "column-major-by-row" storage: element(k,n) at (C+n)*S + KB + k.
// lane: n = C + (l&15); K chunks at lh*8 and 16+lh*8.
__device__ __forceinline__ h16_t fragB(const _Float16* p, int S, int C, int KB, int l15, int lh) {
  int o = (C + l15) * S + KB + lh * 8;
  return frag_ld(p, o, o + 16);
}

// ---------------- prep: weight f32->f16 conversion + padded bias gather ----------------
__global__ void winattn_prep(const float* __restrict__ qkv_w, const float* __restrict__ proj_w,
                             const float* __restrict__ bias_table,
                             _Float16* __restrict__ wq, _Float16* __restrict__ wp,
                             float* __restrict__ wb) {
  int t = blockIdx.x * blockDim.x + threadIdx.x;
  int stride = gridDim.x * blockDim.x;
  for (int i = t; i < 768 * 256; i += stride) wq[i] = (_Float16)qkv_w[i];
  for (int i = t; i < 256 * 256; i += stride) wp[i] = (_Float16)proj_w[i];
  // padded bias [8][64][64]: pad cols get -1e30 (softmax weight -> 0)
  for (int i = t; i < 8 * 64 * 64; i += stride) {
    int h = i >> 12;
    int r = i & 4095;
    int m = r >> 6, n = r & 63;
    float v;
    if (m < 49 && n < 49) {
      int ri = m / 7, ci = m % 7, rj = n / 7, cj = n % 7;
      int idx = (ri - rj + 6) * 13 + (ci - cj + 6);
      v = bias_table[idx * 8 + h];
    } else {
      v = (n >= 49) ? -1e30f : 0.0f;
    }
    wb[i] = v;
  }
}

// ---------------- fused attention kernel ----------------
// LDS layout (halves):
//   sm_x  @ 0      : 64 x 264   (x as f16 A-matrix; later reused as O)
//   sm_q  @ 16896  : 64 x 264   (q, pre-scaled)
//   sm_k  @ 33792  : 64 x 264   (k, row-major token x dim)
//   sm_vt @ 50688  : 256 x 72   (v transposed: [dim][token])
//   sm_p  @ 69120  : 8 x (64 x 72)  (per-wave softmax probs)
#define LDS_HALVES 105984

__global__ __launch_bounds__(256) void winattn_main(
    const float* __restrict__ x, const float* __restrict__ qkv_b,
    const float* __restrict__ proj_b, const _Float16* __restrict__ wq,
    const _Float16* __restrict__ wp, const float* __restrict__ wb,
    float* __restrict__ out) {
  extern __shared__ _Float16 sm[];
  _Float16* sm_x = sm;
  _Float16* sm_q = sm + 16896;
  _Float16* sm_k = sm + 33792;
  _Float16* sm_vt = sm + 50688;
  _Float16* sm_p = sm + 69120;

  const int bw = blockIdx.x;
  const int tid = threadIdx.x;
  const int lane = tid & 31;
  const int wv = tid >> 5;  // wave id 0..7 == head id
  const int l15 = lane & 15;
  const int lh = lane >> 4;
  const float SCALE = 0.17677669529663687f;  // 32^-0.5

  // --- zero pad rows 49..63 of sm_x ---
  {
    h4_t z = {(_Float16)0, (_Float16)0, (_Float16)0, (_Float16)0};
    for (int i = tid; i < 15 * 66; i += 256) {
      int r = 49 + i / 66;
      int c = (i % 66) * 4;
      *(h4_t*)(sm_x + r * 264 + c) = z;
    }
  }
  // --- load x (49x256 f32) -> sm_x f16 ---
  {
    const float* xp = x + (size_t)bw * (49 * 256);
    for (int i = tid; i < 49 * 64; i += 256) {
      int r = i >> 6;
      int c = (i & 63) * 4;
      float4 v = *(const float4*)(xp + r * 256 + c);
      h4_t hv = {(_Float16)v.x, (_Float16)v.y, (_Float16)v.z, (_Float16)v.w};
      *(h4_t*)(sm_x + r * 264 + c) = hv;
    }
  }
  __syncthreads();

  // --- GEMM1: qkv[64,768] = x[64,256] @ qkv_w^T, +bias; q scaled; v stored transposed ---
  for (int g = 0; g < 2; ++g) {
    f8_t acc[3][4];
#pragma unroll
    for (int i = 0; i < 3; ++i)
#pragma unroll
      for (int tm = 0; tm < 4; ++tm) {
        f8_t z = {0.f, 0.f, 0.f, 0.f, 0.f, 0.f, 0.f, 0.f};
        acc[i][tm] = z;
      }
    for (int k0 = 0; k0 < 256; k0 += 32) {
      h16_t aF[4];
#pragma unroll
      for (int tm = 0; tm < 4; ++tm) aF[tm] = fragA(sm_x, 264, tm * 16, k0, l15, lh);
#pragma unroll
      for (int i = 0; i < 3; ++i) {
        int j0 = (wv + (g * 3 + i) * 8) * 16;
        h16_t bF = fragB(wq, 256, j0, k0, l15, lh);
#pragma unroll
        for (int tm = 0; tm < 4; ++tm) acc[i][tm] = WMMA16(aF[tm], bF, acc[i][tm]);
      }
    }
#pragma unroll
    for (int i = 0; i < 3; ++i) {
      int j0 = (wv + (g * 3 + i) * 8) * 16;
      int j = j0 + l15;
      float qb = qkv_b[j];
      int which = j >> 8;  // 0=q 1=k 2=v (uniform per lane across rows)
      float scl = (which == 0) ? SCALE : 1.0f;
#pragma unroll
      for (int tm = 0; tm < 4; ++tm)
#pragma unroll
        for (int r = 0; r < 8; ++r) {
          int m = tm * 16 + r + lh * 8;
          _Float16 hv = (_Float16)((acc[i][tm][r] + qb) * scl);
          if (which == 0)
            sm_q[m * 264 + j] = hv;
          else if (which == 1)
            sm_k[m * 264 + (j - 256)] = hv;
          else
            sm_vt[(j - 512) * 72 + m] = hv;
        }
    }
  }
  __syncthreads();

  // --- GEMM2: S = q @ k^T (head = wv), K = HEAD_DIM = 32 -> single WMMA k-step ---
  {
    f8_t s[4][4];
#pragma unroll
    for (int tm = 0; tm < 4; ++tm)
#pragma unroll
      for (int tn = 0; tn < 4; ++tn) {
        f8_t z = {0.f, 0.f, 0.f, 0.f, 0.f, 0.f, 0.f, 0.f};
        s[tm][tn] = z;
      }
    h16_t qF[4], kF[4];
#pragma unroll
    for (int tm = 0; tm < 4; ++tm) qF[tm] = fragA(sm_q, 264, tm * 16, wv * 32, l15, lh);
#pragma unroll
    for (int tn = 0; tn < 4; ++tn) kF[tn] = fragB(sm_k, 264, tn * 16, wv * 32, l15, lh);
#pragma unroll
    for (int tm = 0; tm < 4; ++tm)
#pragma unroll
      for (int tn = 0; tn < 4; ++tn) s[tm][tn] = WMMA16(qF[tm], kF[tn], s[tm][tn]);

    // + relative position bias (padded, -1e30 in pad columns)
    const float* bh = wb + (wv << 12);
#pragma unroll
    for (int tm = 0; tm < 4; ++tm)
#pragma unroll
      for (int tn = 0; tn < 4; ++tn)
#pragma unroll
        for (int r = 0; r < 8; ++r)
          s[tm][tn][r] += bh[(tm * 16 + r + lh * 8) * 64 + tn * 16 + l15];

    // --- softmax over n (row m lives in one 16-lane half, 4 register tiles) ---
#pragma unroll
    for (int tm = 0; tm < 4; ++tm)
#pragma unroll
      for (int r = 0; r < 8; ++r) {
        float v0 = s[tm][0][r], v1 = s[tm][1][r], v2 = s[tm][2][r], v3 = s[tm][3][r];
        float mx = fmaxf(fmaxf(v0, v1), fmaxf(v2, v3));
        for (int d = 1; d < 16; d <<= 1) mx = fmaxf(mx, __shfl_xor(mx, d));
        float e0 = __expf(v0 - mx), e1 = __expf(v1 - mx);
        float e2 = __expf(v2 - mx), e3 = __expf(v3 - mx);
        float su = e0 + e1 + e2 + e3;
        for (int d = 1; d < 16; d <<= 1) su += __shfl_xor(su, d);
        float inv = 1.0f / su;
        s[tm][0][r] = e0 * inv;
        s[tm][1][r] = e1 * inv;
        s[tm][2][r] = e2 * inv;
        s[tm][3][r] = e3 * inv;
      }

    // --- P -> LDS (f16, wave-private buffer; same-wave LDS ops are in-order) ---
    _Float16* pbuf = sm_p + wv * 4608;  // 64 x 72
#pragma unroll
    for (int tm = 0; tm < 4; ++tm)
#pragma unroll
      for (int tn = 0; tn < 4; ++tn)
#pragma unroll
        for (int r = 0; r < 8; ++r)
          pbuf[(tm * 16 + r + lh * 8) * 72 + tn * 16 + l15] = (_Float16)s[tm][tn][r];

    // --- GEMM3: O_head[64,32] = P[64,64] @ v[64,32], K = 64 -> 2 k-steps ---
    f8_t o[4][2];
#pragma unroll
    for (int tm = 0; tm < 4; ++tm)
#pragma unroll
      for (int tn = 0; tn < 2; ++tn) {
        f8_t z = {0.f, 0.f, 0.f, 0.f, 0.f, 0.f, 0.f, 0.f};
        o[tm][tn] = z;
      }
    for (int ks = 0; ks < 64; ks += 32) {
      h16_t aP[4];
#pragma unroll
      for (int tm = 0; tm < 4; ++tm) aP[tm] = fragA(pbuf, 72, tm * 16, ks, l15, lh);
#pragma unroll
      for (int tn = 0; tn < 2; ++tn) {
        h16_t vF = fragB(sm_vt, 72, wv * 32 + tn * 16, ks, l15, lh);
#pragma unroll
        for (int tm = 0; tm < 4; ++tm) o[tm][tn] = WMMA16(aP[tm], vF, o[tm][tn]);
      }
    }
    // store O to sm_x (x no longer needed) as f16 A-matrix for the proj GEMM
#pragma unroll
    for (int tm = 0; tm < 4; ++tm)
#pragma unroll
      for (int tn = 0; tn < 2; ++tn)
#pragma unroll
        for (int r = 0; r < 8; ++r)
          sm_x[(tm * 16 + r + lh * 8) * 264 + wv * 32 + tn * 16 + l15] =
              (_Float16)o[tm][tn][r];
  }
  __syncthreads();

  // --- GEMM4: out[64,256] = O[64,256] @ proj_w^T + proj_b ---
  {
    f8_t acc[2][4];
#pragma unroll
    for (int t = 0; t < 2; ++t)
#pragma unroll
      for (int tm = 0; tm < 4; ++tm) {
        f8_t z = {0.f, 0.f, 0.f, 0.f, 0.f, 0.f, 0.f, 0.f};
        acc[t][tm] = z;
      }
    for (int k0 = 0; k0 < 256; k0 += 32) {
      h16_t aF[4];
#pragma unroll
      for (int tm = 0; tm < 4; ++tm) aF[tm] = fragA(sm_x, 264, tm * 16, k0, l15, lh);
#pragma unroll
      for (int t = 0; t < 2; ++t) {
        int j0 = (wv * 2 + t) * 16;
        h16_t bF = fragB(wp, 256, j0, k0, l15, lh);
#pragma unroll
        for (int tm = 0; tm < 4; ++tm) acc[t][tm] = WMMA16(aF[tm], bF, acc[t][tm]);
      }
    }
#pragma unroll
    for (int t = 0; t < 2; ++t) {
      int j0 = (wv * 2 + t) * 16;
      float pb = proj_b[j0 + l15];
#pragma unroll
      for (int tm = 0; tm < 4; ++tm)
#pragma unroll
        for (int r = 0; r < 8; ++r) {
          int m = tm * 16 + r + lh * 8;
          if (m < 49)
            out[((size_t)bw * 49 + m) * 256 + j0 + l15] = acc[t][tm][r] + pb;
        }
    }
  }
}

extern "C" void kernel_launch(void* const* d_in, const int* in_sizes, int n_in,
                              void* d_out, int out_size, void* d_ws, size_t ws_size,
                              hipStream_t stream) {
  const float* x = (const float*)d_in[0];
  const float* qkv_w = (const float*)d_in[1];
  const float* qkv_b = (const float*)d_in[2];
  const float* proj_w = (const float*)d_in[3];
  const float* proj_b = (const float*)d_in[4];
  const float* bias_table = (const float*)d_in[5];
  float* out = (float*)d_out;

  // workspace: f16 qkv_w (384KB) | f16 proj_w (128KB) | padded f32 bias (128KB)
  _Float16* wq = (_Float16*)d_ws;
  _Float16* wp = wq + 768 * 256;
  float* wb = (float*)(wp + 256 * 256);

  winattn_prep<<<256, 256, 0, stream>>>(qkv_w, proj_w, bias_table, wq, wp, wb);

  int B_ = in_sizes[0] / (49 * 256);  // 8192
  size_t lds_bytes = (size_t)LDS_HALVES * sizeof(_Float16);  // ~207 KB (<320KB/WGP)
  winattn_main<<<B_, 256, lds_bytes, stream>>>(x, qkv_b, proj_b, wq, wp, wb, out);
}
